// neko_scaling_deform_conv_3x3_65309272703336
// MI455X (gfx1250) — matrix operand order repaired and checked by reading, not compiled, not run on hardware
//
#include <hip/hip_runtime.h>

// Problem constants (from reference setup_inputs)
#define BATCH 4
#define CIN   64
#define COUT  64
#define HH    128
#define WW    128
#define KTAPS 9
#define KTOT  (CIN * KTAPS)   // 576
#define NCHUNK (KTOT / 32)    // 18 chunks of K=32 for wmma_f32_16x16x32_bf16

typedef __attribute__((ext_vector_type(16))) __bf16       v16bf;
typedef __attribute__((ext_vector_type(8)))  float        v8f;
typedef __attribute__((ext_vector_type(4)))  unsigned int u32x4;
typedef __attribute__((ext_vector_type(4)))  int          i32x4;
typedef __attribute__((ext_vector_type(8)))  int          i32x8;

union V16 { u32x4 u[2]; v16bf v; };

struct Corner { float w00, w01, w10, w11; int i00, i01, i10, i11; };  // 32 B

// Use the Tensor Data Mover when the toolchain exposes it. This toolchain uses
// the 6-arg form: (uint32x4 g0, int32x8 g1, int32x4, int32x4, int32x8, i32 cpol).
#if defined(__has_builtin)
# if __has_builtin(__builtin_amdgcn_tensor_load_to_lds) && __has_builtin(__builtin_amdgcn_s_wait_tensorcnt)
#  define USE_TDM 1
# endif
#endif
#ifndef USE_TDM
# define USE_TDM 0
#endif

#if USE_TDM
// Issue a TDM DMA of one 4 KB weight chunk (512 x 8-byte elements, 1-D tile)
// from global memory into LDS. D# layout per CDNA5 ISA ch.8.
__device__ __forceinline__ void tdm_load_b_chunk(const __bf16* gsrc, unsigned lds_off) {
  unsigned long long ga = (unsigned long long)(size_t)gsrc;
  u32x4 g0;
  g0[0] = 1u;                                  // count=1, is_restore=0, gather off
  g0[1] = lds_off;                             // lds_addr (bytes)
  g0[2] = (unsigned)(ga & 0xFFFFFFFFu);        // global_addr[31:0]
  g0[3] = (unsigned)((ga >> 32) & 0x1FFFFFFu)  // global_addr[56:32]
        | (2u << 30);                          // type = 2 ("image")
  i32x8 g1;
  g1[0] = (int)(3u << 16);                     // wg_mask=0, data_size=3 (8 B)
  g1[1] = (int)(512u << 16);                   // tensor_dim0 = 512 (bits 63:48)
  g1[2] = (int)(1u << 16);                     // tensor_dim0 hi=0, tensor_dim1 = 1
  g1[3] = (int)(512u << 16);                   // tile_dim0 = 512 (bits 127:112)
  g1[4] = 1;                                   // tile_dim1 = 1, tile_dim2 = 0
  g1[5] = 512;                                 // tensor_dim0_stride = 512
  g1[6] = 0;
  g1[7] = 0;
  i32x4 z4 = {0, 0, 0, 0};
  i32x8 z8 = {0, 0, 0, 0, 0, 0, 0, 0};
  __builtin_amdgcn_tensor_load_to_lds(g0, g1, z4, z4, z8, 0);
}
#endif

// ---------------------------------------------------------------------------
// Kernel 0: reshuffle dconv_w (O,C,3,3) f32 into a bf16 B-matrix laid out so
// each wave32 lane's WMMA B-fragment is two contiguous b128 loads:
//   bws[(ch*COUT + o)*32 + k] = bf16( w[o][c][tap] ),  kk = ch*32+k = c*9+tap
// ---------------------------------------------------------------------------
__global__ __launch_bounds__(256) void neko_prep_b_kernel(
    const float* __restrict__ dconv_w, __bf16* __restrict__ bws) {
  int idx = blockIdx.x * 256 + threadIdx.x;
  if (idx >= NCHUNK * COUT * 32) return;
  int k   = idx & 31;
  int o   = (idx >> 5) & 63;
  int ch  = idx >> 11;
  int kk  = ch * 32 + k;           // 0..575
  int c   = kk / 9;
  int tap = kk - 9 * c;
  bws[idx] = (__bf16)dconv_w[(o * CIN + c) * 9 + tap];
}

// ---------------------------------------------------------------------------
// Main fused kernel: one workgroup per (b, h) row of 128 pixels.
// 256 threads = 8 waves; wave wv owns pixels [wv*16, wv*16+16) (M=16) and all
// 64 output channels as four 16x16 WMMA accumulator tiles. Double-buffered
// LDS for the bilinear A-tiles and TDM-staged B chunks; one barrier per chunk.
// ---------------------------------------------------------------------------
__global__ __launch_bounds__(256) void neko_deform_conv_wmma_kernel(
    const float* __restrict__ x,
    const float* __restrict__ off_w,
    const float* __restrict__ off_b,
    const float* __restrict__ bn_gamma,
    const float* __restrict__ bn_beta,
    const float* __restrict__ bn_mean,
    const float* __restrict__ bn_var,
    const __bf16* __restrict__ bws,
    const float* __restrict__ dconv_b,
    float* __restrict__ out) {
  __shared__ float  s_alpha[18];
  __shared__ float  s_delta[18];
  __shared__ __align__(16) Corner s_corner[WW][KTAPS];       // 36 KB (reused as
                                                             // output stage later)
  __shared__ __align__(16) __bf16 s_a[2][WW][32];            // 16 KB ping-pong A
  __shared__ __align__(16) __bf16 s_b[2][COUT * 32];         // 8 KB ping-pong B

  const int tid = threadIdx.x;
  const int b   = blockIdx.x >> 7;
  const int h   = blockIdx.x & 127;

  // Fold BN + conv-bias into per-offset-channel affine: tanh(pre*alpha+delta)*0.75
  if (tid < 18) {
    float sc = bn_gamma[tid] * rsqrtf(bn_var[tid] + 1e-5f);
    s_alpha[tid] = sc;
    s_delta[tid] = (off_b[tid] - bn_mean[tid]) * sc + bn_beta[tid];
  }
  __syncthreads();

  // Phase 1: per-pixel offsets (1x1 conv, K=64) -> bilinear corner tables.
  if (tid < WW) {
    const int w = tid;
    float pre[18];
#pragma unroll
    for (int o = 0; o < 18; ++o) pre[o] = 0.f;
    const float* xp = x + (((size_t)b * CIN) * HH + h) * WW + w;
    for (int c = 0; c < CIN; ++c) {
      float xv = xp[(size_t)c * HH * WW];
#pragma unroll
      for (int o = 0; o < 18; ++o) pre[o] = fmaf(xv, off_w[o * CIN + c], pre[o]);
    }
#pragma unroll
    for (int tap = 0; tap < 9; ++tap) {
      float dy = tanhf(fmaf(pre[2 * tap],     s_alpha[2 * tap],     s_delta[2 * tap]))     * 0.75f;
      float dx = tanhf(fmaf(pre[2 * tap + 1], s_alpha[2 * tap + 1], s_delta[2 * tap + 1])) * 0.75f;
      float py = (float)(h + tap / 3 - 1) + dy;
      float px = (float)(w + tap % 3 - 1) + dx;
      float y0f = floorf(py), x0f = floorf(px);
      float wy = py - y0f, wx = px - x0f;
      int y0 = (int)y0f, x0 = (int)x0f;
      int y1 = y0 + 1,   x1 = x0 + 1;
      float vy0 = (y0 >= 0 && y0 < HH) ? 1.f : 0.f;
      float vy1 = (y1 >= 0 && y1 < HH) ? 1.f : 0.f;
      float vx0 = (x0 >= 0 && x0 < WW) ? 1.f : 0.f;
      float vx1 = (x1 >= 0 && x1 < WW) ? 1.f : 0.f;
      int y0c = min(max(y0, 0), HH - 1), y1c = min(max(y1, 0), HH - 1);
      int x0c = min(max(x0, 0), WW - 1), x1c = min(max(x1, 0), WW - 1);
      Corner t;
      t.w00 = (1.f - wy) * (1.f - wx) * vy0 * vx0;
      t.w01 = (1.f - wy) * wx         * vy0 * vx1;
      t.w10 = wy         * (1.f - wx) * vy1 * vx0;
      t.w11 = wy         * wx         * vy1 * vx1;
      t.i00 = y0c * WW + x0c;  t.i01 = y0c * WW + x1c;
      t.i10 = y1c * WW + x0c;  t.i11 = y1c * WW + x1c;
      s_corner[w][tap] = t;
    }
  }

  const int lane  = tid & 31;
  const int wave  = tid >> 5;
  const int m0    = wave * 16;         // first pixel of this wave's M-tile
  const int lhalf = lane >> 4;         // 0: lanes 0-15, 1: lanes 16-31
  const int lmod  = lane & 15;
  const float* xb = x + (size_t)b * CIN * HH * WW;
  const int m  = tid & 127;            // pixel this thread samples for
  const int j0 = (tid >> 7) * 16;      // which 16 K-values within the chunk

  // Cooperative bilinear sampling of one 128x32 bf16 A-chunk into buffer `buf`.
  auto build_chunk = [&](int ch, int buf) {
#pragma unroll
    for (int i = 0; i < 16; ++i) {
      int k   = ch * 32 + j0 + i;
      int c   = k / 9;
      int tap = k - 9 * c;
      const Corner t = s_corner[m][tap];
      const float* p = xb + (size_t)c * HH * WW;
      float v = p[t.i00] * t.w00 + p[t.i01] * t.w01 +
                p[t.i10] * t.w10 + p[t.i11] * t.w11;
      s_a[buf][m][j0 + i] = (__bf16)v;
    }
  };

  // Stage one 4 KB B chunk into LDS buffer `buf`.
  auto stage_b = [&](int ch, int buf) {
#if USE_TDM
    if (wave == 0)
      tdm_load_b_chunk(bws + (size_t)ch * COUT * 32,
                       (unsigned)(size_t)&s_b[buf][0]);
#else
    const u32x4* src = (const u32x4*)(bws + (size_t)ch * COUT * 32);
    ((u32x4*)&s_b[buf][0])[tid] = src[tid];   // 256 threads x 16 B = 4 KB
#endif
  };
  auto wait_b = [&]() {
#if USE_TDM
    if (wave == 0) __builtin_amdgcn_s_wait_tensorcnt(0);
#endif
  };

  // Prologue: corner tables must be complete before sampling, so sync first;
  // then DMA B chunk 0 while building A chunk 0.
  __syncthreads();
  stage_b(0, 0);
  build_chunk(0, 0);
  wait_b();
  __syncthreads();

  // Phase 2: K-loop, one barrier per chunk; DMA + sampling overlap the WMMAs.
  v8f acc[4];
#pragma unroll
  for (int nt = 0; nt < 4; ++nt) acc[nt] = (v8f){};

  for (int ch = 0; ch < NCHUNK; ++ch) {
    const int cur = ch & 1, nxt = cur ^ 1;
    if (ch + 1 < NCHUNK) stage_b(ch + 1, nxt);

    // A fragment (16-bit A 16x32 layout): lanes 0-15 take K 0-7 & 16-23,
    // lanes 16-31 take K 8-15 & 24-31, row = M = pixel-in-tile.
    V16 af;
    {
      const int row = m0 + lmod;
      const int ka  = lhalf * 8;
      af.u[0] = *(const u32x4*)&s_a[cur][row][ka];
      af.u[1] = *(const u32x4*)&s_a[cur][row][ka + 16];
    }
    const int kb = lhalf * 8;   // B: lanes 0-15 hold K 0-7 & 16-23, lanes 16-31 K 8-15 & 24-31

    if (ch + 1 < NCHUNK) build_chunk(ch + 1, nxt);

#pragma unroll
    for (int nt = 0; nt < 4; ++nt) {
      V16 bfrag;
      const __bf16* bp = &s_b[cur][(nt * 16 + lmod) * 32 + kb];
      bfrag.u[0] = *(const u32x4*)bp;
      bfrag.u[1] = *(const u32x4*)(bp + 16);
      acc[nt] = __builtin_amdgcn_wmma_f32_16x16x32_bf16(
          /*neg_a=*/false, af.v, /*neg_b=*/false, bfrag.v,
          /*c_mod=*/(short)0, acc[nt], /*reuse_a=*/false, /*reuse_b=*/false);
    }
    wait_b();
    __syncthreads();
  }

  // Phase 3: transpose D tiles through LDS (reusing the dead corner table,
  // padded stride 132 to dodge bank conflicts), then fully coalesced b128
  // stores: out[b][o][h][0..127] contiguous in w.
  float* s_ostage = (float*)&s_corner[0][0];   // 64*132*4 = 33 KB <= 36 KB
#pragma unroll
  for (int nt = 0; nt < 4; ++nt) {
    const int o = nt * 16 + lmod;
    const int w0 = m0 + lhalf * 8;
#pragma unroll
    for (int r = 0; r < 8; ++r) s_ostage[o * 132 + w0 + r] = acc[nt][r];
  }
  __syncthreads();
  {
    const int o    = tid >> 2;          // 64 output channels
    const int wseg = (tid & 3) * 32;    // 4 segments of 32 pixels
    const float bias = dconv_b[o];
    const float* sp = s_ostage + o * 132 + wseg;
    float* op = out + (((size_t)b * COUT + o) * HH + h) * WW + wseg;
#pragma unroll
    for (int j = 0; j < 8; ++j) {
      float4 v = *(const float4*)(sp + 4 * j);
      float4 r; r.x = v.x + bias; r.y = v.y + bias; r.z = v.z + bias; r.w = v.w + bias;
      *(float4*)(op + 4 * j) = r;
    }
  }
}

extern "C" void kernel_launch(void* const* d_in, const int* in_sizes, int n_in,
                              void* d_out, int out_size, void* d_ws, size_t ws_size,
                              hipStream_t stream) {
  const float* x        = (const float*)d_in[0];
  const float* off_w    = (const float*)d_in[1];
  const float* off_b    = (const float*)d_in[2];
  const float* bn_gamma = (const float*)d_in[3];
  const float* bn_beta  = (const float*)d_in[4];
  const float* bn_mean  = (const float*)d_in[5];
  const float* bn_var   = (const float*)d_in[6];
  const float* dconv_w  = (const float*)d_in[7];
  const float* dconv_b  = (const float*)d_in[8];
  float*  out = (float*)d_out;
  __bf16* bws = (__bf16*)d_ws;   // 18*64*32 bf16 = 72 KB of scratch

  const int nb = (NCHUNK * COUT * 32 + 255) / 256;
  neko_prep_b_kernel<<<nb, 256, 0, stream>>>(dconv_w, bws);
  neko_deform_conv_wmma_kernel<<<BATCH * HH, 256, 0, stream>>>(
      x, off_w, off_b, bn_gamma, bn_beta, bn_mean, bn_var, bws, dconv_b, out);
}
